// LSTMDirectionModel_24086176596736
// MI455X (gfx1250) — matrix operand order repaired and checked by reading, not compiled, not run on hardware
//
#include <hip/hip_runtime.h>

// LSTM(input=3, hidden=32, seq=24, batch=131072) + Linear(32,16)+ReLU+Linear(16,1)
// Recurrent matmul + FC1 on v_wmma_f32_16x16x32_f16 (K=32 == HIDDEN).
// One wave32 owns 16 batch rows; W_hh^T register-resident as 8 f16 B-tiles.
// All LDS staging is wave-private -> only compiler-level wave barriers needed.
// sched_barrier fences bound accumulator liveness; VGPRs capped at 256 (no
// s_set_vgpr_msb). Activations use v_exp_f32 + v_rcp_f32 (TRANS pipe) instead
// of IEEE division expansions.

#define INPUT   3
#define HID     32
#define SEQT    24
#define NWAVES  8     // waves per block
#define MTILE   16    // batch rows per wave

typedef __attribute__((ext_vector_type(16))) _Float16 v16h;
typedef __attribute__((ext_vector_type(8)))  _Float16 v8h;
typedef __attribute__((ext_vector_type(8)))  float    v8f;

__device__ __forceinline__ float sigf(float x) {
    // 1/(1+exp(-x)) with single v_exp_f32 + v_rcp_f32 (both TRANS, co-exec w/ WMMA)
    return __builtin_amdgcn_rcpf(1.0f + __expf(-x));
}
__device__ __forceinline__ float tanh_fast(float x) {
    // tanh(x) = 2/(1+exp(-2x)) - 1
    return __builtin_amdgcn_rcpf(1.0f + __expf(-2.0f * x)) * 2.0f - 1.0f;
}

__global__ __launch_bounds__(256, 1)
__attribute__((amdgpu_num_vgpr(256)))
void lstm_wmma_kernel(
    const float* __restrict__ x,
    const float* __restrict__ w_ih,
    const float* __restrict__ w_hh,
    const float* __restrict__ b_ih,
    const float* __restrict__ b_hh,
    const float* __restrict__ w1,
    const float* __restrict__ b1,
    const float* __restrict__ w2,
    const float* __restrict__ b2,
    float* __restrict__ out)
{
    // per-wave tiles: h transpose buffer (1KB) + x staging (256B). Wave-private!
    __shared__ __align__(16) _Float16 hbuf[NWAVES][16][32];
    __shared__ __align__(16) float    xbuf[NWAVES][16][4];

    const int lane  = threadIdx.x & 31;
    const int wave  = threadIdx.x >> 5;
    const int lhalf = lane & 15;     // column / row-in-tile index
    const int hi    = lane >> 4;     // half-wave select
    const int rowBase = (blockIdx.x * NWAVES + wave) * MTILE;  // grid sized exactly

    // ---- W_hh^T as 8 register-resident f16 B-tiles (32x16 each) --------------
    // B tiles use contiguous k (same order as A), K=32: lane l holds column N = 16n + (l&15); halves are
    // K = (l<16 ? 0..15 : 16..31) packed in pairs.
    const int k0 = hi * 16;
    v16h bt[8];
#pragma unroll
    for (int n = 0; n < 8; ++n) {
        const float* src = w_hh + (size_t)(n * 16 + lhalf) * HID + k0;
        v16h v;
#pragma unroll
        for (int k = 0; k < 16; ++k) v[k] = (_Float16)src[k];
        bt[n] = v;
    }
    // FC1 weight as one more B-tile (32x16): column N = p (FC unit)
    v16h bw1;
    {
        const float* src = w1 + (size_t)lhalf * HID + k0;
#pragma unroll
        for (int k = 0; k < 16; ++k) bw1[k] = (_Float16)src[k];
    }

    // fused bias + input weights for this lane's 8 gate columns
    float bias[8], wih[8][3];
#pragma unroll
    for (int n = 0; n < 8; ++n) {
        const int c = n * 16 + lhalf;
        bias[n] = b_ih[c] + b_hh[c];
#pragma unroll
        for (int k = 0; k < 3; ++k) wih[n][k] = w_ih[c * INPUT + k];
    }

    // cell state: lane covers rows m = r + hi*8 (r=0..7) at j = lhalf and 16+lhalf
    float cst[8][2];
#pragma unroll
    for (int r = 0; r < 8; ++r) { cst[r][0] = 0.f; cst[r][1] = 0.f; }

    v16h aReg = {};   // h0 == 0 in A layout

#pragma unroll 1
    for (int t = 0; t < SEQT; ++t) {
        // stage x tile: lanes 0..15 fetch the 3 features of their row
        if (hi == 0) {
            const float* xp = x + ((size_t)(rowBase + lhalf) * SEQT + t) * INPUT;
            xbuf[wave][lhalf][0] = xp[0];
            xbuf[wave][lhalf][1] = xp[1];
            xbuf[wave][lhalf][2] = xp[2];
            xbuf[wave][lhalf][3] = 0.f;
        }
        __builtin_amdgcn_wave_barrier();   // wave-internal LDS RAW: DS ops are in-order

        // process gate columns in two halves (jj = low/high 16 hidden units);
        // sched_barrier keeps each half's accumulators/temps from overlapping.
#pragma unroll
        for (int jj = 0; jj < 2; ++jj) {
            float4 xq[8];
#pragma unroll
            for (int r = 0; r < 8; ++r)
                xq[r] = *(const float4*)&xbuf[wave][r + hi * 8][0];

            v8f acc[4];
            // C init = bias + x @ W_ih^T (C layout: VGPR r -> row r+hi*8)
#pragma unroll
            for (int g = 0; g < 4; ++g) {
                const int n = g * 2 + jj;          // tile index: gate g, half jj
                v8f a;
#pragma unroll
                for (int r = 0; r < 8; ++r)
                    a[r] = bias[n] + xq[r].x * wih[n][0] + xq[r].y * wih[n][1]
                                   + xq[r].z * wih[n][2];
                acc[g] = a;
            }

            // gates += h @ W_hh^T : 4 x v_wmma_f32_16x16x32_f16 per half
#pragma unroll
            for (int g = 0; g < 4; ++g) {
                acc[g] = __builtin_amdgcn_wmma_f32_16x16x32_f16(
                    false, aReg, false, bt[g * 2 + jj], (short)0, acc[g], false, false);
            }

            // elementwise LSTM update; acc[0..3] = i, f, g, o for this half
#pragma unroll
            for (int r = 0; r < 8; ++r) {
                const float iv = sigf(acc[0][r]);
                const float fv = sigf(acc[1][r]);
                const float gv = tanh_fast(acc[2][r]);
                const float ov = sigf(acc[3][r]);
                const float cn = fv * cst[r][jj] + iv * gv;
                cst[r][jj] = cn;
                const float hn = ov * tanh_fast(cn);
                hbuf[wave][r + hi * 8][jj * 16 + lhalf] = (_Float16)hn;
            }
            __builtin_amdgcn_sched_barrier(0);   // fence: no cross-half code motion
        }
        __builtin_amdgcn_wave_barrier();   // order hbuf stores before A reload

        // reload h in A layout: lane l -> row M=lhalf, K = hi*16..hi*16+15
        const _Float16* hb = &hbuf[wave][lhalf][hi * 16];
        const v8h lo  = *(const v8h*)hb;          // 16B aligned LDS load
        const v8h hi8 = *(const v8h*)(hb + 8);
        aReg = __builtin_shufflevector(lo, hi8,
                 0, 1, 2, 3, 4, 5, 6, 7, 8, 9, 10, 11, 12, 13, 14, 15);
        __builtin_amdgcn_wave_barrier();   // keep next iter's LDS writes below reads
    }

    // ---- head: hid = relu(h @ W1^T + b1); logit = hid . w2 + b2 --------------
    v8f zc = {};
    v8f hid = __builtin_amdgcn_wmma_f32_16x16x32_f16(
        false, aReg, false, bw1, (short)0, zc, false, false);

    const float b1v = b1[lhalf];
    const float w2v = w2[lhalf];
    const float b2v = b2[0];
#pragma unroll
    for (int r = 0; r < 8; ++r) {
        const float hv = fmaxf(hid[r] + b1v, 0.f);
        float val = hv * w2v;
#pragma unroll
        for (int s = 1; s < 16; s <<= 1)           // reduce across the 16 cols
            val += __shfl_xor(val, s, 32);
        if (lhalf == 0)
            out[rowBase + r + hi * 8] = val + b2v;
    }
}

extern "C" void kernel_launch(void* const* d_in, const int* in_sizes, int n_in,
                              void* d_out, int out_size, void* d_ws, size_t ws_size,
                              hipStream_t stream) {
    const float* x    = (const float*)d_in[0];
    const float* w_ih = (const float*)d_in[1];
    const float* w_hh = (const float*)d_in[2];
    const float* b_ih = (const float*)d_in[3];
    const float* b_hh = (const float*)d_in[4];
    const float* w1   = (const float*)d_in[5];
    const float* b1   = (const float*)d_in[6];
    const float* w2   = (const float*)d_in[7];
    const float* b2   = (const float*)d_in[8];
    float* out = (float*)d_out;

    const int B = in_sizes[0] / (SEQT * INPUT);        // 131072
    const int rowsPerBlock = NWAVES * MTILE;           // 128 (B divides evenly)
    dim3 grid((B + rowsPerBlock - 1) / rowsPerBlock);  // 1024 blocks
    lstm_wmma_kernel<<<grid, 256, 0, stream>>>(x, w_ih, w_hh, b_ih, b_hh,
                                               w1, b1, w2, b2, out);
}
